// TTM_Linear_module_3977139716815
// MI455X (gfx1250) — matrix operand order repaired and checked by reading, not compiled, not run on hardware
//
#include <hip/hip_runtime.h>
#include <hip/hip_bf16.h>
#include <stdint.h>

// ---------------------------------------------------------------------------
// TTM_Linear == clip(x @ M, 0, 6), M = dense 1024x1024 product of TT cores.
//   k0: Xbf = bf16(x)                      (one-shot, memory bound)
//   k1..k3: build Mt[out][in] in bf16 (n-major) from the 4 TT cores
//   k4: out = clip(Xbf @ Mt^T, 0, 6) via v_wmma_f32_16x16x32_bf16 with
//       global_load_async_to_lds_b128 double-buffered fills (ASYNCcnt).
// Roofline: ~200 MB HBM traffic (~8.6us @ 23.3 TB/s); 34.4 GFLOP bf16 WMMA;
// Xbf (33.5MB) + Mt (2MB) stay L2-resident (192MB) across the 8 N-passes.
// Async LDS fills remove VGPR staging -> no scratch spills in the WMMA loop.
// ---------------------------------------------------------------------------

typedef __bf16 bf16;
typedef __attribute__((ext_vector_type(16))) __bf16 v16bf;
typedef __attribute__((ext_vector_type(8)))  float  v8f;

#define M_ROWS 16384
#define K_DIM  1024
#define N_DIM  1024
#define BK     32
#define NT     (K_DIM / BK)
#define SA     40   // padded LDS row stride in bf16 elements (80B, 16B aligned)

// workspace layout (bytes)
#define T2_OFF  0                        // 2048 f32   =   8 KB
#define T3_OFF  8192                     // 131072 f32 = 512 KB
#define MT_OFF  (8192 + 524288)          // 1M bf16    =   2 MB
#define XBF_OFF (MT_OFF + 2097152)       // 16.7M bf16 = 33.5 MB (16B aligned)

union FragU { uint4 u[2]; v16bf v; };

__device__ __forceinline__ uint32_t pack_bf16(float lo, float hi) {
  uint32_t a = __float_as_uint(lo);
  uint32_t b = __float_as_uint(hi);
  a += 0x7FFFu + ((a >> 16) & 1u);   // round-to-nearest-even
  b += 0x7FFFu + ((b >> 16) & 1u);
  return (a >> 16) | (b & 0xFFFF0000u);
}

// low 32 bits of a flat shared pointer == LDS byte offset (ISA aperture rule)
__device__ __forceinline__ uint32_t lds_addr32(const void* p) {
  return (uint32_t)(uint64_t)p;
}

__device__ __forceinline__ void async_copy16(uint32_t lds, const void* gptr) {
  asm volatile("global_load_async_to_lds_b128 %0, %1, off"
               :: "v"(lds), "v"((uint64_t)gptr) : "memory");
}

// ---- k0: Xbf = bf16(x); 8 floats per thread -------------------------------
__global__ void ttm_cvt_x(const float* __restrict__ X, bf16* __restrict__ Xbf) {
  size_t idx = (size_t)(blockIdx.x * blockDim.x + threadIdx.x) * 8;
  const float4* p = (const float4*)(X + idx);
  float4 f0 = p[0], f1 = p[1];
  uint4 o;
  o.x = pack_bf16(f0.x, f0.y); o.y = pack_bf16(f0.z, f0.w);
  o.z = pack_bf16(f1.x, f1.y); o.w = pack_bf16(f1.z, f1.w);
  *(uint4*)(Xbf + idx) = o;
}

// ---- k1: T2[E1,B1,r2] = sum_r1 W0[e0][b0*8+r1] * W1[r1*4+e1][b1*8+r2] ------
__global__ void ttm_build_t2(const float* __restrict__ W0,
                             const float* __restrict__ W1,
                             float* __restrict__ T2) {
  int idx = blockIdx.x * blockDim.x + threadIdx.x;   // 0..2047
  int r2 = idx & 7;
  int B1 = (idx >> 3) & 15;
  int E1 = idx >> 7;
  int e1 = E1 & 3, e0 = E1 >> 2;
  int b1 = B1 & 3, b0 = B1 >> 2;
  float s = 0.f;
#pragma unroll
  for (int r1 = 0; r1 < 8; ++r1)
    s += W0[e0 * 32 + b0 * 8 + r1] * W1[(r1 * 4 + e1) * 32 + b1 * 8 + r2];
  T2[idx] = s;
}

// ---- k2: T3[E,B,r3] = sum_r2 T2[E1,B1,r2] * W2[r2*8+e2][b2*8+r3] -----------
__global__ void ttm_build_t3(const float* __restrict__ T2,
                             const float* __restrict__ W2,
                             float* __restrict__ T3) {
  int idx = blockIdx.x * blockDim.x + threadIdx.x;   // 0..131071
  int r3 = idx & 7;
  int B  = (idx >> 3) & 127;
  int E  = idx >> 10;
  int e2 = E & 7, E1 = E >> 3;
  int b2 = B & 7, B1 = B >> 3;
  float s = 0.f;
#pragma unroll
  for (int r2 = 0; r2 < 8; ++r2)
    s += T2[(E1 * 16 + B1) * 8 + r2] * W2[(r2 * 8 + e2) * 64 + b2 * 8 + r3];
  T3[idx] = s;
}

// ---- k3: Mt[n][k] = sum_r3 T3[E,B,r3] * W3[r3*8+e3][b3]  (bf16, n-major) ---
__global__ void ttm_build_mt(const float* __restrict__ T3,
                             const float* __restrict__ W3,
                             bf16* __restrict__ Mt) {
  int idx = blockIdx.x * blockDim.x + threadIdx.x;   // 0..(1M-1)
  int k = idx & 1023;
  int n = idx >> 10;
  int e3 = n & 7, E = n >> 3;
  int b3 = k & 7, B = k >> 3;
  float s = 0.f;
#pragma unroll
  for (int r3 = 0; r3 < 8; ++r3)
    s += T3[(E * 128 + B) * 8 + r3] * W3[(r3 * 8 + e3) * 8 + b3];
  Mt[idx] = (bf16)s;
}

// ---- k4: GEMM out[m][n] = clip( sum_k Xbf[m][k] * Mt[n][k], 0, 6 ) ---------
// block tile 128x128, 8 wave32 (2x4 wave grid, 64x32 per wave = 4x2 WMMA accs),
// BK=32, double-buffered LDS filled by global_load_async_to_lds_b128.
__global__ void __launch_bounds__(256)
ttm_gemm(const bf16* __restrict__ Xbf, const bf16* __restrict__ Mt,
         float* __restrict__ Out) {
  __shared__ bf16 As[2][128 * SA];   // 20480 B
  __shared__ bf16 Bs[2][128 * SA];   // 20480 B  (total 40 KB of 320 KB/WGP)

  const int tid  = threadIdx.x;
  const int lane = tid & 31;
  const int wave = tid >> 5;
  const int wm = wave >> 2;          // 0..1  -> 64-row strip
  const int wn = wave & 3;           // 0..3  -> 32-col strip
  const int lr = lane & 15;          // lane row/col within 16
  const int lh = lane >> 4;          // lane half

  const int m_blk = blockIdx.x * 128;
  const int n_blk = blockIdx.y * 128;

  // tile-fill mapping: 2 threads per row, 16 bf16 (32B = 2x b128) each
  const int frow  = tid >> 1;        // 0..127
  const int fhalf = tid & 1;         // 0/1

  const bf16* ag = Xbf + (size_t)(m_blk + frow) * K_DIM + fhalf * 16;
  const bf16* bg = Mt  + (size_t)(n_blk + frow) * K_DIM + fhalf * 16;

  const uint32_t fill_off = (uint32_t)(frow * SA + fhalf * 16);
  uint32_t ldsA0 = lds_addr32(As[0] + fill_off);
  uint32_t ldsA1 = lds_addr32(As[1] + fill_off);
  uint32_t ldsB0 = lds_addr32(Bs[0] + fill_off);
  uint32_t ldsB1 = lds_addr32(Bs[1] + fill_off);

  v8f acc[4][2];
#pragma unroll
  for (int i = 0; i < 4; ++i)
#pragma unroll
    for (int j = 0; j < 2; ++j)
      acc[i][j] = (v8f){0.f, 0.f, 0.f, 0.f, 0.f, 0.f, 0.f, 0.f};

  auto issue = [&](int kt, int buf) {
    uint32_t la = buf ? ldsA1 : ldsA0;
    uint32_t lb = buf ? ldsB1 : ldsB0;
    const bf16* pa = ag + kt * BK;
    const bf16* pb = bg + kt * BK;
    async_copy16(la,       pa);       // 16B
    async_copy16(la + 16u, pa + 8);   // 16B
    async_copy16(lb,       pb);
    async_copy16(lb + 16u, pb + 8);
  };

  issue(0, 0);

#pragma unroll 2
  for (int kt = 0; kt < NT; ++kt) {
    const int buf = kt & 1;
    asm volatile("s_wait_asynccnt 0x0" ::: "memory");  // my fills of buf landed
    __syncthreads();                                   // everyone's landed

    if (kt + 1 < NT) issue(kt + 1, buf ^ 1);           // overlap with compute

    // B fragments (32x16 bf16): lane = N col (lr); K = lh*16 .. lh*16+15
    FragU bfg[2];
#pragma unroll
    for (int ni = 0; ni < 2; ++ni) {
      const bf16* s = Bs[buf] + (wn * 32 + ni * 16 + lr) * SA + lh * 16;
      bfg[ni].u[0] = *(const uint4*)(s);
      bfg[ni].u[1] = *(const uint4*)(s + 8);
    }
    // A fragment (16x32 bf16) per mi: lane = M row (lr); K = lh*16 .. lh*16+15
#pragma unroll
    for (int mi = 0; mi < 4; ++mi) {
      FragU af;
      const bf16* s = As[buf] + (wm * 64 + mi * 16 + lr) * SA + lh * 16;
      af.u[0] = *(const uint4*)(s);
      af.u[1] = *(const uint4*)(s + 8);
#pragma unroll
      for (int ni = 0; ni < 2; ++ni)
        acc[mi][ni] = __builtin_amdgcn_wmma_f32_16x16x32_bf16(
            false, af.v, false, bfg[ni].v, (short)0, acc[mi][ni],
            false, false);
    }
  }

  // epilogue: C/D layout = lanes 0-15 rows r, lanes 16-31 rows 8+r; N = lr
#pragma unroll
  for (int mi = 0; mi < 4; ++mi) {
    const int m0 = m_blk + wm * 64 + mi * 16 + lh * 8;
#pragma unroll
    for (int ni = 0; ni < 2; ++ni) {
      const int n = n_blk + wn * 32 + ni * 16 + lr;
#pragma unroll
      for (int r = 0; r < 8; ++r) {
        float v = acc[mi][ni][r];
        v = fminf(fmaxf(v, 0.f), 6.f);   // ReLUN(6)
        Out[(size_t)(m0 + r) * N_DIM + n] = v;
      }
    }
  }
}

extern "C" void kernel_launch(void* const* d_in, const int* in_sizes, int n_in,
                              void* d_out, int out_size, void* d_ws, size_t ws_size,
                              hipStream_t stream) {
  const float* x  = (const float*)d_in[0];
  const float* W0 = (const float*)d_in[1];
  const float* W1 = (const float*)d_in[2];
  const float* W2 = (const float*)d_in[3];
  const float* W3 = (const float*)d_in[4];
  float* out = (float*)d_out;

  float* T2  = (float*)((char*)d_ws + T2_OFF);
  float* T3  = (float*)((char*)d_ws + T3_OFF);
  bf16*  Mt  = (bf16*)((char*)d_ws + MT_OFF);
  bf16*  Xbf = (bf16*)((char*)d_ws + XBF_OFF);

  ttm_cvt_x   <<<8192, 256, 0, stream>>>(x, Xbf);   // 16.7M elems, 8/thread
  ttm_build_t2<<<8,    256, 0, stream>>>(W0, W1, T2);
  ttm_build_t3<<<512,  256, 0, stream>>>(T2, W2, T3);
  ttm_build_mt<<<4096, 256, 0, stream>>>(T3, W3, Mt);

  dim3 grid(M_ROWS / 128, N_DIM / 128);   // 128 x 8 blocks
  ttm_gemm<<<grid, 256, 0, stream>>>(Xbf, Mt, out);
}
